// MomentumLSTM_2379411882568
// MI455X (gfx1250) — matrix operand order repaired and checked by reading, not compiled, not run on hardware
//
#include <hip/hip_runtime.h>
#include <hip/hip_bf16.h>

// ---------------------------------------------------------------------------
// MomentumLSTM for MI455X (gfx1250):
//   bf16 WMMA GEMMs with TDM (tensor_load_to_lds) double-buffered staging,
//   fused fp32 momentum scan + LSTM gate activations.
// ---------------------------------------------------------------------------

typedef __attribute__((ext_vector_type(16))) __bf16       v16bf;
typedef __attribute__((ext_vector_type(8)))  float        v8f;
typedef __attribute__((ext_vector_type(4)))  unsigned int v4u;
typedef __attribute__((ext_vector_type(4)))  int          v4i;
typedef __attribute__((ext_vector_type(8)))  int          v8i;

static constexpr int T_ = 512;
static constexpr int B_ = 64;
static constexpr int I_ = 256;
static constexpr int H_ = 512;
static constexpr int O_ = 256;
static constexpr int G_ = 4 * H_;      // 2048
static constexpr int M1 = T_ * B_;     // 32768

static constexpr float MU_ = 0.6f;
static constexpr float S_  = 0.6f;

// GEMM tiling
static constexpr int BM = 128, BN = 128, BK = 32, PADE = 8;  // pad in bf16 elems

// ---------------------------------------------------------------------------
// fp32 -> bf16 elementwise conversion (one-time pass per operand)
// ---------------------------------------------------------------------------
__global__ __launch_bounds__(256)
void f32_to_bf16_kernel(const float* __restrict__ in, __bf16* __restrict__ out,
                        int n4) {
    const int i = blockIdx.x * 256 + threadIdx.x;
    if (i < n4) {
        const float4 v = ((const float4*)in)[i];
        __bf16* o = out + 4 * (size_t)i;
        o[0] = (__bf16)v.x;
        o[1] = (__bf16)v.y;
        o[2] = (__bf16)v.z;
        o[3] = (__bf16)v.w;
    }
}

// ---------------------------------------------------------------------------
// h_proj[b][g] = dot(hx[b,:], W_hh[g,:]) + b_hh[g]      (B x 4H, tiny GEMV)
// ---------------------------------------------------------------------------
__global__ __launch_bounds__(256)
void hproj_kernel(const float* __restrict__ hx, const float* __restrict__ W_hh,
                  const float* __restrict__ b_hh, float* __restrict__ hproj) {
    const int idx = blockIdx.x * 256 + threadIdx.x;   // 0 .. B*G-1
    const int b = idx >> 11;           // / G (2048)
    const int g = idx & (G_ - 1);
    const float4* wr = (const float4*)(W_hh + (size_t)g * H_);
    const float4* xr = (const float4*)(hx   + (size_t)b * H_);
    float s = b_hh[g];
#pragma unroll 4
    for (int k = 0; k < H_ / 4; ++k) {
        float4 w = wr[k];
        float4 x = xr[k];
        s += w.x * x.x + w.y * x.y + w.z * x.z + w.w * x.w;
    }
    hproj[idx] = s;
}

// ---------------------------------------------------------------------------
// TDM: issue a 2D tile load (tile_dim0 = BK elems contiguous, tile_dim1 = 128
// rows, row stride = K elems) from a row-major bf16 tensor into LDS with
// padding: 4 DWORDs (16B) inserted after every 16 DWORDs (64B) streamed,
// matching an LDS layout of [128][BK + PADE] bf16.
// Descriptor packing per CDNA5 ISA.
// ---------------------------------------------------------------------------
__device__ __forceinline__ void tdm_load_tile_2d(const __bf16* gsrc,
                                                 unsigned int lds_byte_off,
                                                 int K_elems, int rows_total) {
    const unsigned long long ga = (unsigned long long)(uintptr_t)gsrc;

    v4u g0;
    g0[0] = 1u;                                    // count=1, user mode, no gather
    g0[1] = lds_byte_off;                          // lds_addr
    g0[2] = (unsigned int)(ga & 0xFFFFFFFFu);      // global_addr[31:0]
    g0[3] = (unsigned int)((ga >> 32) & 0x01FFFFFFu) | (2u << 30); // [56:32]+type=2

    v8i g1;
    // data_size=1 (2B), pad_enable=1, pad_interval=3 (16 DW), pad_amount=3 (4 DW)
    g1[0] = (1 << 16) | (1 << 20) | (3 << 22) | (3 << 25);
    g1[1] = (K_elems & 0xFFFF) << 16;                              // tensor_dim0 lo
    g1[2] = ((unsigned)K_elems >> 16) | ((rows_total & 0xFFFF) << 16); // td0 hi, td1 lo
    g1[3] = ((unsigned)rows_total >> 16) | (BK << 16);             // td1 hi, tile_dim0
    g1[4] = BM;                                                    // tile_dim1, tile_dim2=0
    g1[5] = K_elems;                                               // tensor_dim0_stride lo
    g1[6] = 0;                                                     // stride hi / dim1_stride lo
    g1[7] = 0;

    const v4i z4 = {0, 0, 0, 0};
#if __clang_major__ >= 23
    const v8i z8 = {0, 0, 0, 0, 0, 0, 0, 0};
    __builtin_amdgcn_tensor_load_to_lds(g0, g1, z4, z4, z8, 0);
#else
    __builtin_amdgcn_tensor_load_to_lds(g0, g1, z4, z4, 0);
#endif
}

// ---------------------------------------------------------------------------
// C(MxN) = A(MxK) * W(NxK)^T + bias(N): bf16 operands, fp32 WMMA accumulate.
// Block: 256 threads = 8 waves; tile 128x128x32; TDM double-buffered staging.
// Waves arranged 2(m) x 4(n); each wave: 64x32 patch = 4x2 16x16 accumulators.
// M % 128 == 0, N % 128 == 0, K % 32 == 0 required (true for all our calls).
// ---------------------------------------------------------------------------
template <typename OT>
__global__ __launch_bounds__(256)
void gemm_bias_wmma_tdm(const __bf16* __restrict__ A, const __bf16* __restrict__ W,
                        const float* __restrict__ bias, OT* __restrict__ Cout,
                        int M, int N, int K) {
    __shared__ __bf16 As[2][BM][BK + PADE];   // 2 x 10 KB
    __shared__ __bf16 Bs[2][BN][BK + PADE];   // 2 x 10 KB

    const int tid  = threadIdx.x;
    const int wave = tid >> 5;
    const int lane = tid & 31;
    const int wm   = wave >> 2;     // 0..1
    const int wn   = wave & 3;      // 0..3
    const int m0   = blockIdx.y * BM;
    const int n0   = blockIdx.x * BN;

    v8f acc[4][2];
#pragma unroll
    for (int mt = 0; mt < 4; ++mt)
#pragma unroll
        for (int nt = 0; nt < 2; ++nt)
            acc[mt][nt] = {};

    const unsigned int ldsA0 = (unsigned int)(uintptr_t)(void*)&As[0][0][0];
    const unsigned int ldsA1 = (unsigned int)(uintptr_t)(void*)&As[1][0][0];
    const unsigned int ldsB0 = (unsigned int)(uintptr_t)(void*)&Bs[0][0][0];
    const unsigned int ldsB1 = (unsigned int)(uintptr_t)(void*)&Bs[1][0][0];

    // Prologue: wave 0 kicks off tile 0 into buffer 0.
    if (wave == 0) {
        tdm_load_tile_2d(A + (size_t)m0 * K, ldsA0, K, M);
        tdm_load_tile_2d(W + (size_t)n0 * K, ldsB0, K, N);
    }

    const int fr = lane & 15;             // fragment row/col within 16
    const int hf = lane >> 4;             // lane-half

    const int ksteps = K / BK;
    for (int ki = 0; ki < ksteps; ++ki) {
        const int buf = ki & 1;
        // All waves done reading buf^1 from iteration ki-1 before TDM refills it.
        __syncthreads();
        if (wave == 0) {
            if (ki + 1 < ksteps) {
                const int knext = (ki + 1) * BK;
                tdm_load_tile_2d(A + (size_t)m0 * K + knext,
                                 buf ? ldsA0 : ldsA1, K, M);
                tdm_load_tile_2d(W + (size_t)n0 * K + knext,
                                 buf ? ldsB0 : ldsB1, K, N);
                __builtin_amdgcn_s_wait_tensorcnt(2);  // tile ki complete
            } else {
                __builtin_amdgcn_s_wait_tensorcnt(0);  // last tile complete
            }
        }
        __syncthreads();   // tile ki visible to all waves

        // A fragment (16x32 bf16): lane l holds M=l&15; VGPR0-3 -> K = hf*8+0..7,
        // VGPR4-7 -> K = 16 + hf*8 + 0..7   (CDNA5 ISA 7.12.2)
        v16bf af[4];
#pragma unroll
        for (int mt = 0; mt < 4; ++mt) {
            const __bf16* base = &As[buf][wm * 64 + mt * 16 + fr][0];
#pragma unroll
            for (int e = 0; e < 8; ++e) af[mt][e] = base[hf * 8 + e];
#pragma unroll
            for (int e = 0; e < 8; ++e) af[mt][8 + e] = base[16 + hf * 8 + e];
        }
        // B fragment (32x16 bf16): lane l holds N=l&15; VGPR0-3 -> K = hf*8+0..7,
        // VGPR4-7 -> K = 16 + hf*8 + 0..7
        v16bf bfr[2];
#pragma unroll
        for (int nt = 0; nt < 2; ++nt) {
            const __bf16* base = &Bs[buf][wn * 32 + nt * 16 + fr][0];
#pragma unroll
            for (int e = 0; e < 8; ++e) bfr[nt][e] = base[hf * 8 + e];
#pragma unroll
            for (int e = 0; e < 8; ++e) bfr[nt][8 + e] = base[16 + hf * 8 + e];
        }

#pragma unroll
        for (int mt = 0; mt < 4; ++mt)
#pragma unroll
            for (int nt = 0; nt < 2; ++nt)
                acc[mt][nt] = __builtin_amdgcn_wmma_f32_16x16x32_bf16(
                    false, af[mt], false, bfr[nt], (short)0, acc[mt][nt],
                    false, false);
    }

    // Epilogue: C/D layout — VGPR r: M = r + (lane>>4)*8, N = lane&15
    const int col_l = lane & 15;
    const int row_l = lane >> 4;
#pragma unroll
    for (int mt = 0; mt < 4; ++mt) {
#pragma unroll
        for (int nt = 0; nt < 2; ++nt) {
            const int col = n0 + wn * 32 + nt * 16 + col_l;
            const float bia = bias[col];
            const int rowbase = m0 + wm * 64 + mt * 16 + row_l * 8;
#pragma unroll
            for (int rr = 0; rr < 8; ++rr) {
                const float val = acc[mt][nt][rr] + bia;
                Cout[(size_t)(rowbase + rr) * N + col] = (OT)val;
            }
        }
    }
}

// ---------------------------------------------------------------------------
// Fused momentum scan + gate activations.
// One thread per (b, h): carries 4 fp32 momentum accumulators over T steps.
// xp is bf16 (T,B,4H); writes bf16 hy (T,B,H) and the fp32 tails of d_out.
// ---------------------------------------------------------------------------
__device__ __forceinline__ float sigmoidf_(float x) {
    return 1.0f / (1.0f + __expf(-x));
}

__global__ __launch_bounds__(256)
void scan_act_kernel(const __bf16* __restrict__ xp, const float* __restrict__ hproj,
                     const float* __restrict__ v0, const float* __restrict__ cx,
                     __bf16* __restrict__ hy_bf, float* __restrict__ out) {
    const int idx = blockIdx.x * 256 + threadIdx.x;   // 0 .. B*H-1
    const int b = idx >> 9;             // / H (512)
    const int h = idx & (H_ - 1);

    const size_t vb = (size_t)b * G_;
    float vi = v0[vb + h];
    float vf = v0[vb + H_ + h];
    float vg = v0[vb + 2 * H_ + h];
    float vo = v0[vb + 3 * H_ + h];

    const float pi = hproj[vb + h];
    const float pf = hproj[vb + H_ + h];
    const float pg = hproj[vb + 2 * H_ + h];
    const float po = hproj[vb + 3 * H_ + h];

    const float c0 = cx[(size_t)b * H_ + h];

    float cy = 0.0f, hy = 0.0f;
#pragma unroll 2
    for (int t = 0; t < T_; ++t) {
        const size_t rbase = ((size_t)t * B_ + b) * G_;
        vi = MU_ * vi + S_ * (float)xp[rbase + h];
        vf = MU_ * vf + S_ * (float)xp[rbase + H_ + h];
        vg = MU_ * vg + S_ * (float)xp[rbase + 2 * H_ + h];
        vo = MU_ * vo + S_ * (float)xp[rbase + 3 * H_ + h];

        const float gi = vi + pi;
        const float gf = vf + pf;
        const float gg = vg + pg;
        const float go = vo + po;

        cy = sigmoidf_(gf) * c0 + sigmoidf_(gi) * tanhf(gg);
        hy = sigmoidf_(go) * tanhf(cy);

        hy_bf[((size_t)t * B_ + b) * H_ + h] = (__bf16)hy;
    }

    // d_out layout: [outputs (M1*O)] [hy_last (B*H)] [cy_last (B*H)] [v_last (B*4H)]
    float* out_hy = out + (size_t)M1 * O_;
    float* out_cy = out_hy + (size_t)B_ * H_;
    float* out_v  = out_cy + (size_t)B_ * H_;

    out_hy[(size_t)b * H_ + h] = hy;
    out_cy[(size_t)b * H_ + h] = cy;
    out_v[vb + h]            = vi;
    out_v[vb + H_ + h]       = vf;
    out_v[vb + 2 * H_ + h]   = vg;
    out_v[vb + 3 * H_ + h]   = vo;
}

// ---------------------------------------------------------------------------
extern "C" void kernel_launch(void* const* d_in, const int* in_sizes, int n_in,
                              void* d_out, int out_size, void* d_ws, size_t ws_size,
                              hipStream_t stream) {
    const float* input_seq = (const float*)d_in[0];   // (T,B,I)
    const float* hx        = (const float*)d_in[1];   // (B,H)
    const float* cx        = (const float*)d_in[2];   // (B,H)
    const float* v0        = (const float*)d_in[3];   // (B,4H)
    const float* W_ih      = (const float*)d_in[4];   // (4H,I)
    const float* b_ih      = (const float*)d_in[5];   // (4H)
    const float* W_hh      = (const float*)d_in[6];   // (4H,H)
    const float* b_hh      = (const float*)d_in[7];   // (4H)
    const float* fc_w      = (const float*)d_in[8];   // (O,H)
    const float* fc_b      = (const float*)d_in[9];   // (O)

    // workspace carve
    char* ws = (char*)d_ws;
    size_t off = 0;
    __bf16* xp     = (__bf16*)(ws + off); off += (size_t)M1 * G_ * sizeof(__bf16);
    __bf16* hy_bf  = (__bf16*)(ws + off); off += (size_t)M1 * H_ * sizeof(__bf16);
    __bf16* a1_bf  = (__bf16*)(ws + off); off += (size_t)M1 * I_ * sizeof(__bf16);
    __bf16* w1_bf  = (__bf16*)(ws + off); off += (size_t)G_ * I_ * sizeof(__bf16);
    __bf16* w2_bf  = (__bf16*)(ws + off); off += (size_t)O_ * H_ * sizeof(__bf16);
    float*  hproj  = (float*)(ws + off);  off += (size_t)B_ * G_ * sizeof(float);

    // 0) one-time bf16 conversions of GEMM operands
    f32_to_bf16_kernel<<<(M1 * I_ / 4 + 255) / 256, 256, 0, stream>>>(
        input_seq, a1_bf, M1 * I_ / 4);
    f32_to_bf16_kernel<<<(G_ * I_ / 4 + 255) / 256, 256, 0, stream>>>(
        W_ih, w1_bf, G_ * I_ / 4);
    f32_to_bf16_kernel<<<(O_ * H_ / 4 + 255) / 256, 256, 0, stream>>>(
        fc_w, w2_bf, O_ * H_ / 4);

    // 1) h_proj (tiny, constant over time)
    hproj_kernel<<<(B_ * G_) / 256, 256, 0, stream>>>(hx, W_hh, b_hh, hproj);

    // 2) GEMM1: x_proj = input_seq * W_ih^T + b_ih   (M=32768, N=2048, K=256)
    {
        dim3 grid(G_ / BN, M1 / BM);
        gemm_bias_wmma_tdm<__bf16><<<grid, 256, 0, stream>>>(
            a1_bf, w1_bf, b_ih, xp, M1, G_, I_);
    }

    // 3) fused scan + activations; writes hy (bf16) and fp32 tails of d_out
    scan_act_kernel<<<(B_ * H_) / 256, 256, 0, stream>>>(
        xp, hproj, v0, cx, hy_bf, (float*)d_out);

    // 4) GEMM2: outputs = hy * fc_w^T + fc_b   (M=32768, N=256, K=512)
    {
        dim3 grid(O_ / BN, M1 / BM);
        gemm_bias_wmma_tdm<float><<<grid, 256, 0, stream>>>(
            hy_bf, w2_bf, fc_b, (float*)d_out, M1, O_, H_);
    }
}
